// DeepSeekMoE_14139032338629
// MI455X (gfx1250) — matrix operand order repaired and checked by reading, not compiled, not run on hardware
//
#include <hip/hip_runtime.h>

#define DEVFN __device__ __forceinline__

typedef __attribute__((ext_vector_type(16))) __bf16 v16bf;
typedef __attribute__((ext_vector_type(8)))  __bf16 v8bf;
typedef __attribute__((ext_vector_type(4)))  __bf16 v4bf;
typedef __attribute__((ext_vector_type(8)))  float  v8f;

constexpr int Hdim = 1024;   // hidden
constexpr int Idim = 2048;   // intermediate
constexpr int Edim = 8;      // routed experts
constexpr int Tdim = 4096;   // tokens (2*2048)

constexpr int LDS_ROWB = 80;                 // 64B data + 16B pad -> bank-conflict-free
constexpr int TILE_B   = 128 * LDS_ROWB;     // 10240 B per 128x32 bf16 tile

// ---------------------------------------------------------------- WMMA core
DEVFN v8f wmma_bf16(v16bf a, v16bf b, v8f c) {
    return __builtin_amdgcn_wmma_f32_16x16x32_bf16(
        false, a, false, b, (short)0, c, false, false);
}

DEVFN v16bf cat8(v8bf lo, v8bf hi) {
    v16bf f;
#pragma unroll
    for (int i = 0; i < 8; ++i) { f[i] = lo[i]; f[8 + i] = hi[i]; }
    return f;
}
// LDS tile rows hold K0..31 contiguously (64B) at stride 80B.
// A pattern (interleaved): half h reads bytes [16h,16h+15] and [32+16h, ...]
DEVFN v16bf fragA_lds(const char* tile, int row, int half) {
    const char* rp = tile + row * LDS_ROWB;
    return cat8(*reinterpret_cast<const v8bf*>(rp + half * 16),
                *reinterpret_cast<const v8bf*>(rp + 32 + half * 16));
}
// B pattern (interleaved): half h reads bytes [16h,16h+15] and [32+16h, ...]
DEVFN v16bf fragB_lds(const char* tile, int row, int half) {
    const char* rp = tile + row * LDS_ROWB + half * 16;
    return cat8(*reinterpret_cast<const v8bf*>(rp),
                *reinterpret_cast<const v8bf*>(rp + 32));
}

// ----------------------------------------------- async global -> LDS (16B/lane)
DEVFN unsigned lds_off(const void* p) {           // flat-LDS addr[31:0] == offset
    return (unsigned)(size_t)p;
}
DEVFN void async_b128(unsigned lds_byte_off, const void* src) {
    asm volatile("global_load_async_to_lds_b128 %0, %1, off"
                 :: "v"(lds_byte_off), "v"(src) : "memory");
}
DEVFN void wait_async0() {
    asm volatile("s_wait_asynccnt 0x0" ::: "memory");
}

// ------------------------------------------------------- fp32 -> bf16 pass
__global__ void cvt_f32_to_bf16(const float* __restrict__ in,
                                __bf16* __restrict__ out, int n) {
    int i = (blockIdx.x * 256 + threadIdx.x) * 4;
    if (i >= n) return;
    const float4 v = *reinterpret_cast<const float4*>(in + i);
    v4bf o;
    o[0] = (__bf16)v.x; o[1] = (__bf16)v.y;
    o[2] = (__bf16)v.z; o[3] = (__bf16)v.w;
    *reinterpret_cast<v4bf*>(out + i) = o;
}

// ---------------------------------------------------------------- router
__global__ void router_topk(const float* __restrict__ x,
                            const float* __restrict__ rw,
                            const float* __restrict__ rb,
                            int* __restrict__ topk_idx,
                            float* __restrict__ topk_w) {
    const int lane = threadIdx.x & 31;
    const int wave = threadIdx.x >> 5;
    const int t = blockIdx.x * 8 + wave;
    if (t >= Tdim) return;
    const float* xr = x + (size_t)t * Hdim;
    float acc[Edim];
#pragma unroll
    for (int e = 0; e < Edim; ++e) acc[e] = 0.f;
    for (int h = lane; h < Hdim; h += 32) {
        float xv = xr[h];
#pragma unroll
        for (int e = 0; e < Edim; ++e) acc[e] += xv * rw[e * Hdim + h];
    }
#pragma unroll
    for (int e = 0; e < Edim; ++e) {
#pragma unroll
        for (int off = 16; off > 0; off >>= 1)
            acc[e] += __shfl_down(acc[e], off, 32);
    }
    if (lane == 0) {
        float p[Edim];
#pragma unroll
        for (int e = 0; e < Edim; ++e)
            p[e] = 1.f / (1.f + __expf(-(acc[e] + rb[e])));
        int i0 = 0; float p0 = p[0];
#pragma unroll
        for (int e = 1; e < Edim; ++e)
            if (p[e] > p0) { p0 = p[e]; i0 = e; }
        int i1 = -1; float p1 = -1.f;
#pragma unroll
        for (int e = 0; e < Edim; ++e)
            if (e != i0 && p[e] > p1) { p1 = p[e]; i1 = e; }
        float s = p0 + p1;
        topk_idx[2 * t]     = i0;
        topk_idx[2 * t + 1] = i1;
        topk_w[2 * t]       = p0 / s;
        topk_w[2 * t + 1]   = p1 / s;
    }
}

// ------------------------------------------------ deterministic list build
__global__ void build_lists(const int* __restrict__ topk_idx,
                            int* __restrict__ counts,
                            int* __restrict__ offsets,
                            int* __restrict__ tokenlist,
                            int* __restrict__ slot_of) {
    const int lane = threadIdx.x & 31;
    const int e = threadIdx.x >> 5;
    __shared__ int cnt[Edim];
    __shared__ int off[Edim + 1];
    int c = 0;
    for (int t0 = 0; t0 < Tdim; t0 += 32) {
        int t = t0 + lane;
        bool m = (topk_idx[2 * t] == e) || (topk_idx[2 * t + 1] == e);
        c += __popc((unsigned)__ballot(m));
    }
    if (lane == 0) cnt[e] = c;
    __syncthreads();
    if (threadIdx.x == 0) {
        int a = 0;
        for (int i = 0; i < Edim; ++i) { off[i] = a; a += cnt[i]; }
        off[Edim] = a;
    }
    __syncthreads();
    if (threadIdx.x < Edim) {
        counts[threadIdx.x]  = cnt[threadIdx.x];
        offsets[threadIdx.x] = off[threadIdx.x];
    }
    int base = off[e];
    for (int t0 = 0; t0 < Tdim; t0 += 32) {
        int t = t0 + lane;
        bool m0 = (topk_idx[2 * t] == e), m1 = (topk_idx[2 * t + 1] == e);
        bool m = m0 || m1;
        unsigned bal = (unsigned)__ballot(m);
        int rank = __popc(bal & ((1u << lane) - 1u));
        if (m) {
            int slot = base + rank;
            tokenlist[slot] = t;
            slot_of[2 * t + (m0 ? 0 : 1)] = slot;
        }
        base += __popc(bal);
    }
}

// ------------------- GEMM 1: h = silu(x Wg^T) * (x Wu^T), LDS-staged async
// 256 threads = 8 waves (4x2). Wave tile 32x64, block tile 128x128.
__global__ __launch_bounds__(256) void gemm_gateup(
    const __bf16* __restrict__ xb,
    const __bf16* __restrict__ wg_all,
    const __bf16* __restrict__ wu_all,
    __bf16* __restrict__ hbuf,
    const int* __restrict__ tokenlist,
    const int* __restrict__ counts,
    const int* __restrict__ offsets) {
    constexpr int SLAB = 3 * TILE_B;          // A | G | U
    __shared__ char lds[2 * SLAB];            // 60 KB
    const int tid  = threadIdx.x;
    const int lane = tid & 31;
    const int wave = tid >> 5;
    const int wm = wave >> 1;                 // 0..3
    const int wn = wave & 1;                  // 0..1
    const int r = lane & 15;
    const int half = lane >> 4;
    const int e = blockIdx.z;

    int Ne, slotBase;
    const __bf16 *wgp, *wup;
    if (tokenlist) {
        Ne = counts[e]; slotBase = offsets[e];
        wgp = wg_all + (size_t)e * Idim * Hdim;
        wup = wu_all + (size_t)e * Idim * Hdim;
    } else {
        Ne = Tdim; slotBase = 0; wgp = wg_all; wup = wu_all;
    }
    if ((int)blockIdx.y * 128 >= Ne) return;
    const int Mblk  = blockIdx.y * 128;
    const int Nbase = blockIdx.x * 128;

    // staging: 512 16B-chunks per 128x32 tile; thread covers rows tid/4, tid/4+64
    const int row0 = tid >> 2, q0 = tid & 3;
    const int row1 = row0 + 64;
    auto arow = [&](int row) -> const __bf16* {
        int mg = Mblk + row;
        int mc = mg < Ne ? mg : Ne - 1;
        int tok = tokenlist ? tokenlist[slotBase + mc] : mc;
        return xb + (size_t)tok * Hdim;
    };
    const __bf16* aS0 = arow(row0) + q0 * 8;
    const __bf16* aS1 = arow(row1) + q0 * 8;
    const __bf16* gS0 = wgp + (size_t)(Nbase + row0) * Hdim + q0 * 8;
    const __bf16* gS1 = wgp + (size_t)(Nbase + row1) * Hdim + q0 * 8;
    const __bf16* uS0 = wup + (size_t)(Nbase + row0) * Hdim + q0 * 8;
    const __bf16* uS1 = wup + (size_t)(Nbase + row1) * Hdim + q0 * 8;
    const unsigned d0 = lds_off(lds) + row0 * LDS_ROWB + q0 * 16;
    const unsigned d1 = lds_off(lds) + row1 * LDS_ROWB + q0 * 16;

    auto stage = [&](int sel, int k0) {
        const unsigned sb = (unsigned)sel * SLAB;
        async_b128(d0 + sb,              aS0 + k0);
        async_b128(d1 + sb,              aS1 + k0);
        async_b128(d0 + sb + TILE_B,     gS0 + k0);
        async_b128(d1 + sb + TILE_B,     gS1 + k0);
        async_b128(d0 + sb + 2 * TILE_B, uS0 + k0);
        async_b128(d1 + sb + 2 * TILE_B, uS1 + k0);
    };

    v8f accG[2][4] = {};
    v8f accU[2][4] = {};
    stage(0, 0);
    const int KT = Hdim / 32;
    for (int kt = 0; kt < KT; ++kt) {
        wait_async0();
        __syncthreads();          // slab[sel] ready; slab[sel^1] free to overwrite
        const int sel = kt & 1;
        if (kt + 1 < KT) stage(sel ^ 1, (kt + 1) * 32);
        const char* slab = lds + sel * SLAB;
        v16bf a[2], gb[4], ub[4];
#pragma unroll
        for (int s = 0; s < 2; ++s)
            a[s] = fragA_lds(slab, wm * 32 + s * 16 + r, half);
#pragma unroll
        for (int tt = 0; tt < 4; ++tt) {
            gb[tt] = fragB_lds(slab + TILE_B,     wn * 64 + tt * 16 + r, half);
            ub[tt] = fragB_lds(slab + 2 * TILE_B, wn * 64 + tt * 16 + r, half);
        }
#pragma unroll
        for (int s = 0; s < 2; ++s) {
#pragma unroll
            for (int tt = 0; tt < 4; ++tt) {
                accG[s][tt] = wmma_bf16(a[s], gb[tt], accG[s][tt]);
                accU[s][tt] = wmma_bf16(a[s], ub[tt], accU[s][tt]);
            }
        }
    }

#pragma unroll
    for (int s = 0; s < 2; ++s) {
#pragma unroll
        for (int tt = 0; tt < 4; ++tt) {
            int n = Nbase + wn * 64 + tt * 16 + r;
#pragma unroll
            for (int rv = 0; rv < 8; ++rv) {
                int ml = Mblk + wm * 32 + s * 16 + rv + half * 8;
                if (ml < Ne) {
                    float g = accG[s][tt][rv];
                    float u = accU[s][tt][rv];
                    float sig = __builtin_amdgcn_rcpf(1.f + __expf(-g));
                    hbuf[(size_t)(slotBase + ml) * Idim + n] = (__bf16)(g * sig * u);
                }
            }
        }
    }
}

// ------------------------- GEMM 2: y = h Wd^T, LDS-staged async (bf16 in)
__global__ __launch_bounds__(256) void gemm_down(
    const __bf16* __restrict__ hbuf,
    const __bf16* __restrict__ wd_all,
    float* __restrict__ outp,
    const int* __restrict__ counts,
    const int* __restrict__ offsets) {
    constexpr int SLAB = 2 * TILE_B;          // A | B
    __shared__ char lds[2 * SLAB];            // 40 KB
    const int tid  = threadIdx.x;
    const int lane = tid & 31;
    const int wave = tid >> 5;
    const int wm = wave >> 1;
    const int wn = wave & 1;
    const int r = lane & 15;
    const int half = lane >> 4;
    const int e = blockIdx.z;

    int Ne, rowBase;
    const __bf16* wdp;
    if (counts) {
        Ne = counts[e]; rowBase = offsets[e];
        wdp = wd_all + (size_t)e * Hdim * Idim;
    } else {
        Ne = Tdim; rowBase = 0; wdp = wd_all;
    }
    if ((int)blockIdx.y * 128 >= Ne) return;
    const int Mblk  = blockIdx.y * 128;
    const int Nbase = blockIdx.x * 128;

    const int row0 = tid >> 2, q0 = tid & 3;
    const int row1 = row0 + 64;
    auto arow = [&](int row) -> const __bf16* {
        int mg = Mblk + row;
        int mc = mg < Ne ? mg : Ne - 1;
        return hbuf + (size_t)(rowBase + mc) * Idim;
    };
    const __bf16* aS0 = arow(row0) + q0 * 8;
    const __bf16* aS1 = arow(row1) + q0 * 8;
    const __bf16* bS0 = wdp + (size_t)(Nbase + row0) * Idim + q0 * 8;
    const __bf16* bS1 = wdp + (size_t)(Nbase + row1) * Idim + q0 * 8;
    const unsigned d0 = lds_off(lds) + row0 * LDS_ROWB + q0 * 16;
    const unsigned d1 = lds_off(lds) + row1 * LDS_ROWB + q0 * 16;

    auto stage = [&](int sel, int k0) {
        const unsigned sb = (unsigned)sel * SLAB;
        async_b128(d0 + sb,          aS0 + k0);
        async_b128(d1 + sb,          aS1 + k0);
        async_b128(d0 + sb + TILE_B, bS0 + k0);
        async_b128(d1 + sb + TILE_B, bS1 + k0);
    };

    v8f acc[2][4] = {};
    stage(0, 0);
    const int KT = Idim / 32;
    for (int kt = 0; kt < KT; ++kt) {
        wait_async0();
        __syncthreads();
        const int sel = kt & 1;
        if (kt + 1 < KT) stage(sel ^ 1, (kt + 1) * 32);
        const char* slab = lds + sel * SLAB;
        v16bf a[2], b[4];
#pragma unroll
        for (int s = 0; s < 2; ++s)
            a[s] = fragA_lds(slab, wm * 32 + s * 16 + r, half);
#pragma unroll
        for (int tt = 0; tt < 4; ++tt)
            b[tt] = fragB_lds(slab + TILE_B, wn * 64 + tt * 16 + r, half);
#pragma unroll
        for (int s = 0; s < 2; ++s) {
#pragma unroll
            for (int tt = 0; tt < 4; ++tt)
                acc[s][tt] = wmma_bf16(a[s], b[tt], acc[s][tt]);
        }
    }

#pragma unroll
    for (int s = 0; s < 2; ++s) {
#pragma unroll
        for (int tt = 0; tt < 4; ++tt) {
            int n = Nbase + wn * 64 + tt * 16 + r;
#pragma unroll
            for (int rv = 0; rv < 8; ++rv) {
                int ml = Mblk + wm * 32 + s * 16 + rv + half * 8;
                if (ml < Ne)
                    outp[(size_t)(rowBase + ml) * Hdim + n] = acc[s][tt][rv];
            }
        }
    }
}

// ---------------------------------------------------------------- combine
__global__ void combine(float* __restrict__ out,
                        const float* __restrict__ ybuf,
                        const int* __restrict__ slot_of,
                        const float* __restrict__ topk_w) {
    int idx = blockIdx.x * blockDim.x + threadIdx.x;
    if (idx >= Tdim * Hdim) return;
    int t = idx >> 10;
    int h = idx & (Hdim - 1);
    float v = out[idx];
    v += topk_w[2 * t]     * ybuf[(size_t)slot_of[2 * t]     * Hdim + h];
    v += topk_w[2 * t + 1] * ybuf[(size_t)slot_of[2 * t + 1] * Hdim + h];
    out[idx] = v;
}

// ---------------------------------------------------------------- launch
extern "C" void kernel_launch(void* const* d_in, const int* in_sizes, int n_in,
                              void* d_out, int out_size, void* d_ws, size_t ws_size,
                              hipStream_t stream) {
    (void)in_sizes; (void)n_in; (void)out_size; (void)ws_size;
    const float* x  = (const float*)d_in[0];
    const float* sg = (const float*)d_in[1];
    const float* su = (const float*)d_in[2];
    const float* sd = (const float*)d_in[3];
    const float* rw = (const float*)d_in[4];
    const float* rb = (const float*)d_in[5];
    const float* wg = (const float*)d_in[6];
    const float* wu = (const float*)d_in[7];
    const float* wd = (const float*)d_in[8];
    float* out = (float*)d_out;

    char* ws = (char*)d_ws;
    int*    topk_idx  = (int*)  (ws + 0);
    float*  topk_w    = (float*)(ws + 32768);
    int*    slot_of   = (int*)  (ws + 65536);
    int*    tokenlist = (int*)  (ws + 98304);
    int*    counts    = (int*)  (ws + 131072);
    int*    offsets   = (int*)  (ws + 131072 + 256);

    char* p = ws + 262144;
    auto alloc = [&](size_t bytes) {
        char* r = p;
        p += (bytes + 255) & ~(size_t)255;
        return r;
    };
    __bf16* xb   = (__bf16*)alloc((size_t)Tdim * Hdim * 2);          //  8 MB
    __bf16* sgb  = (__bf16*)alloc((size_t)Idim * Hdim * 2);          //  4 MB
    __bf16* sub  = (__bf16*)alloc((size_t)Idim * Hdim * 2);          //  4 MB
    __bf16* sdb  = (__bf16*)alloc((size_t)Hdim * Idim * 2);          //  4 MB
    __bf16* wgb  = (__bf16*)alloc((size_t)Edim * Idim * Hdim * 2);   // 32 MB
    __bf16* wub  = (__bf16*)alloc((size_t)Edim * Idim * Hdim * 2);   // 32 MB
    __bf16* wdb  = (__bf16*)alloc((size_t)Edim * Hdim * Idim * 2);   // 32 MB
    __bf16* hbuf = (__bf16*)alloc((size_t)2 * Tdim * Idim * 2);      // 32 MB
    float*  ybuf = (float*) alloc((size_t)2 * Tdim * Hdim * 4);      // 32 MB

    auto cvt = [&](const float* src, __bf16* dst, int n) {
        cvt_f32_to_bf16<<<n / (4 * 256), 256, 0, stream>>>(src, dst, n);
    };
    cvt(x,  xb,  Tdim * Hdim);
    cvt(sg, sgb, Idim * Hdim);
    cvt(su, sub, Idim * Hdim);
    cvt(sd, sdb, Hdim * Idim);
    cvt(wg, wgb, Edim * Idim * Hdim);
    cvt(wu, wub, Edim * Idim * Hdim);
    cvt(wd, wdb, Edim * Hdim * Idim);

    router_topk<<<Tdim / 8, 256, 0, stream>>>(x, rw, rb, topk_idx, topk_w);
    build_lists<<<1, 256, 0, stream>>>(topk_idx, counts, offsets, tokenlist, slot_of);
    // shared expert (identity token list); reuses first Tdim rows of hbuf
    gemm_gateup<<<dim3(Idim / 128, Tdim / 128, 1), 256, 0, stream>>>(
        xb, sgb, sub, hbuf, nullptr, nullptr, nullptr);
    gemm_down<<<dim3(Hdim / 128, Tdim / 128, 1), 256, 0, stream>>>(
        hbuf, sdb, out, nullptr, nullptr);
    // routed experts
    gemm_gateup<<<dim3(Idim / 128, Tdim / 128, Edim), 256, 0, stream>>>(
        xb, wgb, wub, hbuf, tokenlist, counts, offsets);
    gemm_down<<<dim3(Hdim / 128, Tdim / 128, Edim), 256, 0, stream>>>(
        hbuf, wdb, ybuf, counts, offsets);
    combine<<<(Tdim * Hdim) / 256, 256, 0, stream>>>(out, ybuf, slot_of, topk_w);
}
